// PatchcoreModel_9586367004799
// MI455X (gfx1250) — matrix operand order repaired and checked by reading, not compiled, not run on hardware
//
#include <hip/hip_runtime.h>
#include <math.h>

typedef __attribute__((ext_vector_type(16))) __bf16 v16bf;
typedef __attribute__((ext_vector_type(8)))  __bf16 v8bf;
typedef __attribute__((ext_vector_type(4)))  __bf16 v4bf;
typedef __attribute__((ext_vector_type(8)))  float  v8f;

#define DIM      1536
#define BM       256
#define BN       128
#define KK       32
#define KSTEPS   (DIM / KK)   // 48
#define LDSTRIDE 40           // bf16 elems per LDS chunk row (32 + pad); 80B rows -> 16B aligned frags
#define SCRS     68           // floats per transpose-scratch row (64 + pad); 272B -> 16B aligned
#define NK       9            // top-k

#define FINF __builtin_inff()

// ---- unsorted top-9 with tracked worst; fully unrolled so t[] stays in VGPRs ----
__device__ __forceinline__ void topk_insert(float (&t)[NK], float &worst, float v) {
  if (v < worst) {
    bool done = false;
#pragma unroll
    for (int i = 0; i < NK; ++i) {
      if (!done && t[i] == worst) { t[i] = v; done = true; }
    }
    worst = t[0];
#pragma unroll
    for (int i = 1; i < NK; ++i) worst = fmaxf(worst, t[i]);
  }
}

// A fragment (16x32 bf16): low lanes K{0..7,16..23}, high lanes K{8..15,24..31}
__device__ __forceinline__ v16bf ld_fragA(const __bf16* S, int row0, int lo, int hi) {
  const __bf16* p = S + (row0 + lo) * LDSTRIDE + hi * 8;
  v8bf a = *(const v8bf*)p;
  v8bf b = *(const v8bf*)(p + 16);
  return __builtin_shufflevector(a, b, 0,1,2,3,4,5,6,7,8,9,10,11,12,13,14,15);
}

// B fragment (32x16 bf16): low lanes K{0..7,16..23}, high lanes K{8..15,24..31}
__device__ __forceinline__ v16bf ld_fragB(const __bf16* S, int row0, int lo, int hi) {
  const __bf16* p = S + (row0 + lo) * LDSTRIDE + hi * 8;
  v8bf a = *(const v8bf*)p;
  v8bf b = *(const v8bf*)(p + 16);
  return __builtin_shufflevector(a, b, 0,1,2,3,4,5,6,7,8,9,10,11,12,13,14,15);
}

// ---------------- fused bf16-WMMA cdist^2 + per-block top-9 ----------------
// Block = 256 rows x 128 cols. 8 waves, each owning 32 rows x 128 cols:
// 2x8 = 16 WMMA tiles -> 128 accumulator VGPRs. 32-bit global offsets keep
// addressing in SADDR+voffset form so allocation stays in the v0-v255 bank
// (4 waves/SIMD -> 2 workgroups resident per WGP).
__global__ __launch_bounds__(256, 2)
void patchcore_gemm_topk(const float* __restrict__ E, const float* __restrict__ MB,
                         const float* __restrict__ x2, const float* __restrict__ y2,
                         float* __restrict__ cand, int rows, int cols, int nbc) {
  // Phase 1 (GEMM): As/Bs bf16 chunks.  Phase 2 (scan): scr overlaps As/Bs (disjoint in time).
  __shared__ __align__(16) char smem[35328];
  __bf16* As  = (__bf16*)smem;            // 256*40*2 = 20480 B
  __bf16* Bs  = (__bf16*)(smem + 20480);  // 128*40*2 = 10240 B (ends 30720)
  float*  scr = (float*)smem;             // 8 waves * 16*68*4 = 34816 B (phase 2 only)
  float*  y2s = (float*)(smem + 34816);   // 512 B

  const int tid  = threadIdx.x;
  const int lane = tid & 31, wave = tid >> 5;   // wave owns rows [wave*32, wave*32+32)
  const int lo   = lane & 15, hi = lane >> 4;
  const int rowBase = blockIdx.y * BM;
  const int colBase = blockIdx.x * BN;

  if (tid < BN) { int c0 = colBase + tid; y2s[tid] = (c0 < cols) ? y2[c0] : 0.f; }

  // 32-bit row offsets (matrices < 128 MB, element offsets fit easily)
  int rowOffA[8];   // (clamped row) * DIM for the 8 A sub-rows this thread loads
  int rowOffB[4];
#pragma unroll
  for (int j = 0; j < 8; ++j) {
    int r  = (tid + 256 * j) >> 3;
    int ga = rowBase + r; if (ga >= rows) ga = rows - 1;
    rowOffA[j] = ga * DIM;
  }
#pragma unroll
  for (int j = 0; j < 4; ++j) {
    int r  = (tid + 256 * j) >> 3;
    int gb = colBase + r; if (gb >= cols) gb = cols - 1;
    rowOffB[j] = gb * DIM;
  }
  const int kpA = ((tid & 7) << 2);   // k offset within chunk for this thread
  const int kpB = ((tid & 7) << 2);

  v8f c[2][8];
#pragma unroll
  for (int rt = 0; rt < 2; ++rt)
#pragma unroll
    for (int ct = 0; ct < 8; ++ct) c[rt][ct] = {};

  for (int bk = 0; bk < KSTEPS; ++bk) {
    const int kb  = bk * KK;
    const int kbn = (bk + 1 < KSTEPS) ? kb + KK : kb;   // next-chunk prefetch target
    __syncthreads();  // protect LDS from previous iteration's readers
    // cooperative load of 256x32 (A) and 128x32 (B) fp32 chunks, converted to bf16 in LDS;
    // prefetch the next chunk's cachelines (no LOADcnt, no registers held)
#pragma unroll
    for (int j = 0; j < 8; ++j) {
      int r = (tid + 256 * j) >> 3;
      float4 va = *(const float4*)(E + rowOffA[j] + kb + kpA);
      __builtin_prefetch(E + rowOffA[j] + kbn + kpA, 0, 3);
      v4bf ba; ba[0] = (__bf16)va.x; ba[1] = (__bf16)va.y; ba[2] = (__bf16)va.z; ba[3] = (__bf16)va.w;
      *(v4bf*)(As + r * LDSTRIDE + kpA) = ba;
    }
#pragma unroll
    for (int j = 0; j < 4; ++j) {
      int r = (tid + 256 * j) >> 3;
      float4 vb = *(const float4*)(MB + rowOffB[j] + kb + kpB);
      __builtin_prefetch(MB + rowOffB[j] + kbn + kpB, 0, 3);
      v4bf bb; bb[0] = (__bf16)vb.x; bb[1] = (__bf16)vb.y; bb[2] = (__bf16)vb.z; bb[3] = (__bf16)vb.w;
      *(v4bf*)(Bs + r * LDSTRIDE + kpB) = bb;
    }
    __syncthreads();

    v16bf af[2], bf[8];
#pragma unroll
    for (int rt = 0; rt < 2; ++rt) af[rt] = ld_fragA(As, wave * 32 + rt * 16, lo, hi);
#pragma unroll
    for (int ct = 0; ct < 8; ++ct) bf[ct] = ld_fragB(Bs, ct * 16, lo, hi);
#pragma unroll
    for (int rt = 0; rt < 2; ++rt)
#pragma unroll
      for (int ct = 0; ct < 8; ++ct)
        c[rt][ct] = __builtin_amdgcn_wmma_f32_16x16x32_bf16(
            false, af[rt], false, bf[ct], (short)0, c[rt][ct], false, false);
  }

  // ---- phase 2: d2 = x2 + y2 - 2*dot, per-row top-9 over this 128-col block ----
#pragma unroll
  for (int rt = 0; rt < 2; ++rt) {
    float t[NK];
#pragma unroll
    for (int i = 0; i < NK; ++i) t[i] = FINF;
    float worst = FINF;

    int row = rowBase + wave * 32 + rt * 16 + lo;
    float myx2 = x2[row < rows ? row : rows - 1];

#pragma unroll
    for (int g = 0; g < 2; ++g) {   // ct groups {0..3},{4..7} -> scr holds 16x64 at a time
      __syncthreads();
#pragma unroll
      for (int ct = 0; ct < 4; ++ct)
#pragma unroll
        for (int vi = 0; vi < 8; ++vi)
          scr[wave * (16 * SCRS) + (vi + 8 * hi) * SCRS + ct * 16 + lo] = c[rt][g * 4 + ct][vi];
      __syncthreads();

      const float* sw = scr + wave * (16 * SCRS) + lo * SCRS + hi * 32;
      int colLoc0 = g * 64 + hi * 32;
#pragma unroll 1            // keep as a real loop: 8x fewer insert sites, t[] stays in VGPRs
      for (int j = 0; j < 8; ++j) {
        float4 v4 = *(const float4*)(sw + j * 4);
        float vv[4] = {v4.x, v4.y, v4.z, v4.w};
#pragma unroll
        for (int e = 0; e < 4; ++e) {
          int cl  = colLoc0 + j * 4 + e;
          int col = colBase + cl;
          float d = (col < cols) ? (myx2 + y2s[cl] - 2.f * vv[e]) : FINF;
          topk_insert(t, worst, d);
        }
      }
    }

    // merge the two lane-halves of each row via shuffle (no LDS, no barrier)
#pragma unroll
    for (int i = 0; i < NK; ++i) {
      float v = __shfl_xor(t[i], 16, 32);
      if (hi == 0) topk_insert(t, worst, v);
    }
    if (hi == 0 && row < rows) {
      float* dst = cand + ((size_t)row * nbc + blockIdx.x) * NK;
#pragma unroll
      for (int i = 0; i < NK; ++i) dst[i] = t[i];
    }
  }
}

// ---------------- exact fp32 squared row norms (one wave per row) ----------------
__global__ void patchcore_rownorm(const float* __restrict__ src, int nrows, float* __restrict__ dst) {
  int w    = (blockIdx.x * blockDim.x + threadIdx.x) >> 5;
  int lane = threadIdx.x & 31;
  if (w >= nrows) return;
  const float4* p = (const float4*)(src + (size_t)w * DIM);
  float s = 0.f;
  for (int j = lane; j < DIM / 4; j += 32) {
    float4 v = p[j];
    s += v.x * v.x + v.y * v.y + v.z * v.z + v.w * v.w;
  }
#pragma unroll
  for (int off = 16; off > 0; off >>= 1) s += __shfl_xor(s, off, 32);
  if (lane == 0) dst[w] = s;
}

// ---------------- merge per-block candidates -> exact top-9 (+ sqrt) ----------------
__global__ void patchcore_merge(const float* __restrict__ cand, int rows, int nbc,
                                float* __restrict__ sc9, float* __restrict__ pmin) {
  int r = blockIdx.x * blockDim.x + threadIdx.x;
  if (r >= rows) return;
  float t[NK];
#pragma unroll
  for (int i = 0; i < NK; ++i) t[i] = FINF;
  float worst = FINF;
  const float* p = cand + (size_t)r * nbc * NK;
  int n = nbc * NK;
#pragma unroll 1
  for (int i = 0; i < n; ++i) topk_insert(t, worst, p[i]);
  float mn = t[0];
#pragma unroll
  for (int i = 1; i < NK; ++i) mn = fminf(mn, t[i]);
#pragma unroll
  for (int i = 0; i < NK; ++i) sc9[(size_t)r * NK + i] = sqrtf(fmaxf(t[i], 0.f));
  pmin[r] = sqrtf(fmaxf(mn, 0.f));
}

// ---------------- anomaly score: argmax + softmax-style reweight ----------------
__global__ void patchcore_score(const float* __restrict__ pmin, const float* __restrict__ sc9,
                                int rows, float* __restrict__ out) {
  __shared__ float sv[256];
  __shared__ int   si[256];
  int t = threadIdx.x;
  float bv = -FINF; int bi = 0;
  for (int r = t; r < rows; r += 256) {
    float v = pmin[r];
    if (v > bv) { bv = v; bi = r; }
  }
  sv[t] = bv; si[t] = bi;
  __syncthreads();
  for (int o = 128; o > 0; o >>= 1) {
    if (t < o) {
      if (sv[t + o] > sv[t] || (sv[t + o] == sv[t] && si[t + o] < si[t])) {
        sv[t] = sv[t + o]; si[t] = si[t + o];
      }
    }
    __syncthreads();
  }
  if (t == 0) {
    int idx = si[0];
    float m = -FINF, s = 0.f;
#pragma unroll
    for (int i = 0; i < NK; ++i) {
      float e = expf(sc9[(size_t)idx * NK + i]);
      m = fmaxf(m, e); s += e;
    }
    out[0] = (1.f - m / s) * sv[0];
  }
}

// ---------------- gaussian weights (sigma=4, ksize=33) ----------------
__global__ void patchcore_gauss(float* __restrict__ g) {
  int i = threadIdx.x;
  if (i < 33) {
    float s = 0.f;
    for (int j = 0; j < 33; ++j) { float x = (float)(j - 16); s += expf(-(x * x) / 32.f); }
    float xi = (float)(i - 16);
    g[i] = expf(-(xi * xi) / 32.f) / s;
  }
}

// ---------------- nearest x8 upsample 28x28 -> 224x224 ----------------
__global__ void patchcore_upsample(const float* __restrict__ pmin, float* __restrict__ up, int npix) {
  int o = blockIdx.x * blockDim.x + threadIdx.x;
  if (o >= npix) return;
  int b = o / (224 * 224);
  int r = o % (224 * 224);
  int y = r / 224, x = r % 224;
  up[o] = pmin[b * 784 + (y >> 3) * 28 + (x >> 3)];
}

// ---------------- separable 33-tap blur with reflect padding ----------------
__global__ void patchcore_blurH(const float* __restrict__ src, const float* __restrict__ g,
                                float* __restrict__ dst, int npix) {
  int o = blockIdx.x * blockDim.x + threadIdx.x;
  if (o >= npix) return;
  int b = o / (224 * 224);
  int r = o % (224 * 224);
  int y = r / 224, x = r % 224;
  const float* row = src + (size_t)b * 224 * 224 + (size_t)y * 224;
  float acc = 0.f;
#pragma unroll 11
  for (int i = 0; i < 33; ++i) {
    int xx = x - 16 + i;
    if (xx < 0) xx = -xx;
    if (xx > 223) xx = 446 - xx;
    acc += g[i] * row[xx];
  }
  dst[o] = acc;
}

__global__ void patchcore_blurV(const float* __restrict__ src, const float* __restrict__ g,
                                float* __restrict__ dst, int npix) {
  int o = blockIdx.x * blockDim.x + threadIdx.x;
  if (o >= npix) return;
  int b = o / (224 * 224);
  int r = o % (224 * 224);
  int y = r / 224, x = r % 224;
  const float* img = src + (size_t)b * 224 * 224;
  float acc = 0.f;
#pragma unroll 11
  for (int i = 0; i < 33; ++i) {
    int yy = y - 16 + i;
    if (yy < 0) yy = -yy;
    if (yy > 223) yy = 446 - yy;
    acc += g[i] * img[(size_t)yy * 224 + x];
  }
  dst[o] = acc;
}

extern "C" void kernel_launch(void* const* d_in, const int* in_sizes, int n_in,
                              void* d_out, int out_size, void* d_ws, size_t ws_size,
                              hipStream_t stream) {
  const float* E  = (const float*)d_in[0];   // [rows x 1536] fp32
  const float* MB = (const float*)d_in[1];   // [cols x 1536] fp32
  int rows = in_sizes[0] / DIM;              // 3136
  int cols = in_sizes[1] / DIM;              // 20000
  int B    = rows / 784;                     // 4
  int nbc  = (cols + BN - 1) / BN;           // 157
  int nbr  = (rows + BM - 1) / BM;           // 13
  int npix = B * 224 * 224;                  // 200704

  // workspace carve-up (~20 MB total)
  char* w = (char*)d_ws;
  auto alloc = [&](size_t bytes) -> char* {
    char* p = w; w += (bytes + 255) & ~(size_t)255; return p;
  };
  float* y2   = (float*)alloc((size_t)cols * 4);
  float* x2   = (float*)alloc((size_t)rows * 4);
  float* g    = (float*)alloc(33 * 4);
  float* cand = (float*)alloc((size_t)rows * nbc * NK * 4);
  float* sc9  = (float*)alloc((size_t)rows * NK * 4);
  float* pmin = (float*)alloc((size_t)rows * 4);
  float* up   = (float*)alloc((size_t)npix * 4);
  float* tmp  = (float*)alloc((size_t)npix * 4);
  (void)ws_size; (void)n_in; (void)out_size;

  float* amap  = (float*)d_out;           // [B,1,224,224] flat
  float* score = amap + (size_t)npix;     // scalar

  patchcore_rownorm<<<(rows * 32 + 255) / 256, 256, 0, stream>>>(E, rows, x2);
  patchcore_rownorm<<<(cols * 32 + 255) / 256, 256, 0, stream>>>(MB, cols, y2);
  patchcore_gauss<<<1, 64, 0, stream>>>(g);

  dim3 grid(nbc, nbr);
  patchcore_gemm_topk<<<grid, 256, 0, stream>>>(E, MB, x2, y2, cand, rows, cols, nbc);

  patchcore_merge<<<(rows + 255) / 256, 256, 0, stream>>>(cand, rows, nbc, sc9, pmin);
  patchcore_score<<<1, 256, 0, stream>>>(pmin, sc9, rows, score);
  patchcore_upsample<<<(npix + 255) / 256, 256, 0, stream>>>(pmin, up, npix);
  patchcore_blurH<<<(npix + 255) / 256, 256, 0, stream>>>(up, g, tmp, npix);
  patchcore_blurV<<<(npix + 255) / 256, 256, 0, stream>>>(tmp, g, amap, npix);
}
